// ParentNodeLayer_61168924229861
// MI455X (gfx1250) — matrix operand order repaired and checked by reading, not compiled, not run on hardware
//
#include <hip/hip_runtime.h>
#include <hip/hip_bf16.h>
#include <math.h>

// ---- problem constants (match reference) ----
#define B_ 8
#define N_ 2048
#define D_ 128
#define H_ 4
#define F_ 64
#define S_ 128
#define HF_ 256   // H*F
#define NEG_ (-1e9f)
#define SLOPE_ 0.2f

typedef __bf16 bf16_t;
typedef __attribute__((ext_vector_type(16))) __bf16 v16bf;
typedef __attribute__((ext_vector_type(8)))  __bf16 v8bf;
typedef __attribute__((ext_vector_type(8)))  float  v8f;

// ---------------------------------------------------------------------------
// WMMA helpers (CDNA5 wave32, D = A(16x32 bf16) * B(32x16 bf16) + C(16x16 f32))
// ---------------------------------------------------------------------------
__device__ __forceinline__ v8f wmma_bf16(v16bf a, v16bf b, v8f c) {
    return __builtin_amdgcn_wmma_f32_16x16x32_bf16(
        /*neg_a=*/false, a, /*neg_b=*/false, b,
        /*c_mod=*/(short)0, c, /*reuse_a=*/false, /*reuse_b=*/false);
}

// A-operand tile 16(M) x 32(K) from row-major bf16, tile origin = base, stride ld.
// Layout (ISA 7.12.2): lane L -> row m=L&15; lanes<16 hold K={0..7,16..23},
// lanes>=16 hold K={8..15,24..31}; element e<8 -> K=kb+e, e>=8 -> K=kb+16+(e-8).
__device__ __forceinline__ v16bf load_a_tile(const bf16_t* base, int ld, int lane) {
    int m  = lane & 15;
    int kb = (lane >> 4) << 3;          // 0 or 8
    const bf16_t* p = base + (size_t)m * ld + kb;
    v8bf lo = *reinterpret_cast<const v8bf*>(p);
    v8bf hi = *reinterpret_cast<const v8bf*>(p + 16);
    v16bf r;
#pragma unroll
    for (int e = 0; e < 8; ++e) { r[e] = lo[e]; r[8 + e] = hi[e]; }
    return r;
}

// B-operand tile 32(K) x 16(N) read from a PRE-TRANSPOSED matrix: baseT is the
// tile origin of a (Ncols x K) row-major buffer (row = output column n).
// Layout: lane L -> col n=L&15; lanes<16 hold K={0..7,16..23}, lanes>=16
// K={8..15,24..31}; element e<8 -> K=kb+e, e>=8 -> K=kb+16+(e-8).
__device__ __forceinline__ v16bf load_b_tile(const bf16_t* baseT, int ld, int lane) {
    int n  = lane & 15;
    int kb = (lane >> 4) << 3;          // 0 or 8
    const bf16_t* p = baseT + (size_t)n * ld + kb;
    v8bf lo = *reinterpret_cast<const v8bf*>(p);
    v8bf hi = *reinterpret_cast<const v8bf*>(p + 16);
    v16bf r;
#pragma unroll
    for (int e = 0; e < 8; ++e) { r[e] = lo[e]; r[8 + e] = hi[e]; }
    return r;
}

__device__ __forceinline__ v8f zero_v8f() {
    v8f z = {0.f, 0.f, 0.f, 0.f, 0.f, 0.f, 0.f, 0.f};
    return z;
}

// ---------------------------------------------------------------------------
// Conversion kernels
// ---------------------------------------------------------------------------
__global__ void cvt_bf16_kernel(const float* __restrict__ src, bf16_t* __restrict__ dst, int n) {
    int i = blockIdx.x * blockDim.x + threadIdx.x;
    if (i < n) dst[i] = (bf16_t)src[i];
}

// src: (nmat, R, C) f32 row-major -> dst: (nmat, C, R) bf16
__global__ void cvt_transpose_kernel(const float* __restrict__ src, bf16_t* __restrict__ dst,
                                     int R, int C, int total) {
    int i = blockIdx.x * blockDim.x + threadIdx.x;
    if (i >= total) return;
    int rc  = R * C;
    int mat = i / rc;
    int rem = i - mat * rc;
    int r = rem / C;
    int c = rem - r * C;
    dst[((size_t)mat * C + c) * R + r] = (bf16_t)src[i];
}

// ---------------------------------------------------------------------------
// Projection: h[b,h,:,:] = X[b] (N x D) @ W_h (D x F), store transposed bf16
// hT layout: (B,H,F,N).  Block = 128 thr (4 waves, wave = head), grid = B*(N/16)
// ---------------------------------------------------------------------------
__global__ void proj_kernel(const bf16_t* __restrict__ inbf,   // (B*N, D) bf16
                            const bf16_t* __restrict__ WT,     // (H, F, D) bf16
                            bf16_t* __restrict__ hT) {         // (B,H,F,N) bf16
    int blk  = blockIdx.x;
    int b    = blk / (N_ / 16);
    int i0   = (blk % (N_ / 16)) * 16;
    int h    = threadIdx.x >> 5;
    int lane = threadIdx.x & 31;

    const bf16_t* Abase = inbf + ((size_t)b * N_ + i0) * D_;
    const bf16_t* Bbase = WT + (size_t)h * F_ * D_;

    v8f acc[4];
#pragma unroll
    for (int ft = 0; ft < 4; ++ft) acc[ft] = zero_v8f();

#pragma unroll
    for (int k0 = 0; k0 < D_; k0 += 32) {
        v16bf a = load_a_tile(Abase + k0, D_, lane);
#pragma unroll
        for (int ft = 0; ft < 4; ++ft) {
            v16bf bt = load_b_tile(Bbase + (size_t)(ft * 16) * D_ + k0, D_, lane);
            acc[ft] = wmma_bf16(a, bt, acc[ft]);
        }
    }

    int half = lane >> 4;
    int n    = lane & 15;
    bf16_t* out = hT + (size_t)(b * H_ + h) * F_ * N_;
#pragma unroll
    for (int ft = 0; ft < 4; ++ft) {
#pragma unroll
        for (int v = 0; v < 8; ++v) {
            int f    = ft * 16 + n;
            int node = i0 + half * 8 + v;
            out[(size_t)f * N_ + node] = (bf16_t)acc[ft][v];
        }
    }
}

// ---------------------------------------------------------------------------
// Edge scores: esrc[b,h,i] = sum_f hT[b,h,f,i]*a_src[h,f] (and edst)
// ---------------------------------------------------------------------------
__global__ void edge_scores_kernel(const bf16_t* __restrict__ hT,
                                   const float* __restrict__ a_src,
                                   const float* __restrict__ a_dst,
                                   float* __restrict__ esrc,
                                   float* __restrict__ edst) {
    int tid = blockIdx.x * blockDim.x + threadIdx.x;  // 0..B*H*N-1
    if (tid >= B_ * H_ * N_) return;
    int i  = tid % N_;
    int bh = tid / N_;
    int h  = bh % H_;
    const bf16_t* base = hT + (size_t)bh * F_ * N_;
    float s = 0.f, d = 0.f;
#pragma unroll 4
    for (int f = 0; f < F_; ++f) {
        float hv = (float)base[(size_t)f * N_ + i];
        s += hv * a_src[h * F_ + f];
        d += hv * a_dst[h * F_ + f];
    }
    esrc[tid] = s;
    edst[tid] = d;
}

// ---------------------------------------------------------------------------
// Flash attention over the dense masked N x N graph.
// Block = 128 thr (wave = head) so each adj row-block streams once for 4 heads.
// P (16x32) is produced directly in WMMA A-operand layout; row stats via
// shfl_xor(16); accumulator C-layout rescale via shfl broadcast.
// attnout: (B, N, H*F) bf16 (heads concatenated -> A operand of epilogue GEMM)
// ---------------------------------------------------------------------------
__global__ void attention_kernel(const float* __restrict__ adj,   // (B,N,N)
                                 const float* __restrict__ t,     // (B,N)
                                 const float* __restrict__ esrc,  // (B,H,N)
                                 const float* __restrict__ edst,  // (B,H,N)
                                 const bf16_t* __restrict__ hT,   // (B,H,F,N)
                                 bf16_t* __restrict__ attnout) {  // (B,N,HF)
    int blk  = blockIdx.x;
    int b    = blk / (N_ / 16);
    int i0   = (blk % (N_ / 16)) * 16;
    int h    = threadIdx.x >> 5;
    int lane = threadIdx.x & 31;
    int m    = lane & 15;
    int half = lane >> 4;
    int i    = i0 + m;
    int bh   = b * H_ + h;

    float esrc_i = esrc[(size_t)bh * N_ + i];
    float t_i    = t[(size_t)b * N_ + i];
    const float*  adjrow = adj + ((size_t)b * N_ + i) * N_;
    const float*  edr    = edst + (size_t)bh * N_;
    const float*  tr     = t + (size_t)b * N_;
    const bf16_t* hTb    = hT + (size_t)bh * F_ * N_;

    float row_max = -INFINITY;
    float row_sum = 0.f;
    v8f acc[4];
#pragma unroll
    for (int ft = 0; ft < 4; ++ft) acc[ft] = zero_v8f();

    int kb = half * 8;   // A-layout K base for this lane

    for (int j0 = 0; j0 < N_; j0 += 32) {
        float pv[16];
        float cmax = -INFINITY;
#pragma unroll
        for (int g = 0; g < 2; ++g) {
            int jb = j0 + kb + g * 16;
#pragma unroll
            for (int u = 0; u < 8; ++u) {
                int j    = jb + u;
                float aj = adjrow[j];
                float ee = esrc_i + edr[j];
                ee = (ee > 0.f) ? ee : SLOPE_ * ee;          // leaky relu
                float dt  = fabsf(t_i - tr[j]);
                float Aij = (dt > 0.f) ? (1.0f / dt) : 0.f;  // time affinity
                ee *= Aij;
                ee = (aj > 0.f) ? ee : NEG_;                 // adjacency mask
                pv[g * 8 + u] = ee;
                cmax = fmaxf(cmax, ee);
            }
        }
        // lanes L and L+16 hold the two K-halves of the same row
        cmax = fmaxf(cmax, __shfl_xor(cmax, 16, 32));
        float nmax  = fmaxf(row_max, cmax);
        float scale = __expf(row_max - nmax);
        row_max = nmax;

        float csum = 0.f;
        v16bf pa;
#pragma unroll
        for (int e = 0; e < 16; ++e) {
            float p = __expf(pv[e] - nmax);
            csum += p;
            pa[e] = (bf16_t)p;
        }
        csum += __shfl_xor(csum, 16, 32);
        row_sum = row_sum * scale + csum;

        // rescale accumulators: row r's scale lives in lane r (r = 0..15)
#pragma unroll
        for (int v = 0; v < 8; ++v) {
            float sr = __shfl(scale, half * 8 + v, 32);
#pragma unroll
            for (int ft = 0; ft < 4; ++ft) acc[ft][v] *= sr;
        }

        // O += P (16x32) @ H (32x64)
#pragma unroll
        for (int ft = 0; ft < 4; ++ft) {
            v16bf bt = load_b_tile(hTb + (size_t)(ft * 16) * N_ + j0, N_, lane);
            acc[ft] = wmma_bf16(pa, bt, acc[ft]);
        }
    }

    // normalize by row_sum and store (B, N, HF) bf16
#pragma unroll
    for (int v = 0; v < 8; ++v) {
        float rs  = __shfl(row_sum, half * 8 + v, 32);
        float inv = 1.0f / rs;
        int node  = i0 + half * 8 + v;
#pragma unroll
        for (int ft = 0; ft < 4; ++ft) {
            int col = h * F_ + ft * 16 + (lane & 15);
            attnout[((size_t)b * N_ + node) * HF_ + col] = (bf16_t)(acc[ft][v] * inv);
        }
    }
}

// ---------------------------------------------------------------------------
// Epilogue GEMM: out (B*N x Dout) = attnout (B*N x 256) @ Wo (256 x Dout)
// WoT is pre-transposed (Dout x 256). Dout = 128.
// Optionally also writes bf16 row-major copy and bf16 transposed copy
// (needed to feed layer-2 projection and x_parent).
// Block = 256 thr (8 waves, wave = output col tile), grid = B*(N/16)
// ---------------------------------------------------------------------------
__global__ void out_proj_kernel(const bf16_t* __restrict__ Abf,  // (B*N, 256)
                                const bf16_t* __restrict__ WoT,  // (128, 256)
                                float* __restrict__ outF,        // (B*N, 128)
                                bf16_t* __restrict__ out_bf,     // (B*N, 128) or null
                                bf16_t* __restrict__ outT,       // (B, 128, N) or null
                                int writeAux) {
    int blk  = blockIdx.x;
    int b    = blk / (N_ / 16);
    int i0   = (blk % (N_ / 16)) * 16;
    int ct   = threadIdx.x >> 5;        // col tile 0..7
    int lane = threadIdx.x & 31;

    const bf16_t* Abase = Abf + ((size_t)b * N_ + i0) * HF_;
    const bf16_t* Bbase = WoT + (size_t)(ct * 16) * HF_;

    v8f acc = zero_v8f();
#pragma unroll
    for (int k0 = 0; k0 < HF_; k0 += 32) {
        v16bf a  = load_a_tile(Abase + k0, HF_, lane);
        v16bf bt = load_b_tile(Bbase + k0, HF_, lane);
        acc = wmma_bf16(a, bt, acc);
    }

    int half = lane >> 4;
    int n    = lane & 15;
#pragma unroll
    for (int v = 0; v < 8; ++v) {
        int node = i0 + half * 8 + v;
        int d    = ct * 16 + n;
        float val = acc[v];
        outF[((size_t)b * N_ + node) * 128 + d] = val;
        if (writeAux) {
            out_bf[((size_t)b * N_ + node) * 128 + d] = (bf16_t)val;
            outT[((size_t)b * 128 + d) * N_ + node]   = (bf16_t)val;
        }
    }
}

// ---------------------------------------------------------------------------
// Softmax over S for assignment logits; writes s (f32, into d_out) and sT bf16.
// Block = 128 thr (one per sidx), grid = B*N.
// ---------------------------------------------------------------------------
__global__ void softmax_s_kernel(const float* __restrict__ logits,  // (B*N, S)
                                 float* __restrict__ s_out,         // (B*N, S)
                                 bf16_t* __restrict__ sT) {         // (B, S, N)
    int node = blockIdx.x;
    int b    = node / N_;
    int n    = node - b * N_;
    int sidx = threadIdx.x;
    float v = logits[(size_t)node * S_ + sidx];

    __shared__ float sh[4];
    float m = v;
#pragma unroll
    for (int off = 16; off > 0; off >>= 1) m = fmaxf(m, __shfl_xor(m, off, 32));
    int wid = threadIdx.x >> 5;
    if ((threadIdx.x & 31) == 0) sh[wid] = m;
    __syncthreads();
    float bm = fmaxf(fmaxf(sh[0], sh[1]), fmaxf(sh[2], sh[3]));
    __syncthreads();

    float p = __expf(v - bm);
    float sum = p;
#pragma unroll
    for (int off = 16; off > 0; off >>= 1) sum += __shfl_xor(sum, off, 32);
    if ((threadIdx.x & 31) == 0) sh[wid] = sum;
    __syncthreads();
    float bs = (sh[0] + sh[1]) + (sh[2] + sh[3]);

    float r = p / bs;
    s_out[(size_t)node * S_ + sidx]         = r;
    sT[((size_t)b * S_ + sidx) * N_ + n]    = (bf16_t)r;
}

// ---------------------------------------------------------------------------
// x_parent[b] (S x D) = sT[b] (S x N) @ z[b] (N x D)   (A = sT, B via zT)
// grid = B * (S/16) M-tiles, block = 256 (8 waves = 8 D col-tiles)
// ---------------------------------------------------------------------------
__global__ void xparent_kernel(const bf16_t* __restrict__ sT,  // (B, S, N)
                               const bf16_t* __restrict__ zT,  // (B, D, N)
                               float* __restrict__ xp) {       // (B, S, D)
    int blk  = blockIdx.x;
    int b    = blk / (S_ / 16);
    int m0   = (blk % (S_ / 16)) * 16;
    int ct   = threadIdx.x >> 5;
    int lane = threadIdx.x & 31;

    const bf16_t* Abase = sT + ((size_t)b * S_ + m0) * N_;
    const bf16_t* Bbase = zT + ((size_t)b * D_ + ct * 16) * N_;

    v8f acc = zero_v8f();
    for (int k0 = 0; k0 < N_; k0 += 32) {
        v16bf a  = load_a_tile(Abase + k0, N_, lane);
        v16bf bt = load_b_tile(Bbase + k0, N_, lane);
        acc = wmma_bf16(a, bt, acc);
    }

    int half = lane >> 4;
    int n    = lane & 15;
#pragma unroll
    for (int v = 0; v < 8; ++v) {
        int row = m0 + half * 8 + v;
        xp[((size_t)b * S_ + row) * D_ + ct * 16 + n] = acc[v];
    }
}

// ---------------------------------------------------------------------------
// t_parent[b, sidx] = sum_n t[b,n] * s[b,n,sidx]
// ---------------------------------------------------------------------------
__global__ void tparent_kernel(const float* __restrict__ t,
                               const float* __restrict__ s,
                               float* __restrict__ tp) {
    int b    = blockIdx.x;
    int sidx = threadIdx.x;
    const float* tb = t + (size_t)b * N_;
    const float* sb = s + (size_t)b * N_ * S_;
    float acc = 0.f;
    for (int n = 0; n < N_; ++n) acc += tb[n] * sb[(size_t)n * S_ + sidx];
    tp[b * S_ + sidx] = acc;
}

// ---------------------------------------------------------------------------
// host launcher
// ---------------------------------------------------------------------------
extern "C" void kernel_launch(void* const* d_in, const int* in_sizes, int n_in,
                              void* d_out, int out_size, void* d_ws, size_t ws_size,
                              hipStream_t stream) {
    (void)in_sizes; (void)n_in; (void)out_size; (void)ws_size;

    const float* x       = (const float*)d_in[0];
    const float* adj     = (const float*)d_in[1];
    const float* t       = (const float*)d_in[2];
    const float* W_e     = (const float*)d_in[3];
    const float* a_src_e = (const float*)d_in[4];
    const float* a_dst_e = (const float*)d_in[5];
    const float* Wo_e    = (const float*)d_in[6];
    const float* W_a     = (const float*)d_in[7];
    const float* a_src_a = (const float*)d_in[8];
    const float* a_dst_a = (const float*)d_in[9];
    const float* Wo_a    = (const float*)d_in[10];

    float* out = (float*)d_out;
    float* xp_out = out;                                   // (B,S,D)
    float* s_out  = out + (size_t)B_ * S_ * D_;            // (B,N,S)
    float* tp_out = s_out + (size_t)B_ * N_ * S_;          // (B,1,S)

    // ---- workspace carve-up ----
    char*  ws  = (char*)d_ws;
    size_t off = 0;
    auto alloc = [&](size_t bytes) -> void* {
        void* p = ws + off;
        off = (off + bytes + 255) & ~(size_t)255;
        return p;
    };
    bf16_t* xbf     = (bf16_t*)alloc((size_t)B_ * N_ * D_ * 2);      // (B*N, D)
    bf16_t* WT_e    = (bf16_t*)alloc((size_t)H_ * F_ * D_ * 2);      // (H, F, D)
    bf16_t* WT_a    = (bf16_t*)alloc((size_t)H_ * F_ * D_ * 2);
    bf16_t* WoT_e   = (bf16_t*)alloc((size_t)D_ * HF_ * 2);          // (128, 256)
    bf16_t* WoT_a   = (bf16_t*)alloc((size_t)S_ * HF_ * 2);          // (128, 256)
    bf16_t* hT      = (bf16_t*)alloc((size_t)B_ * H_ * F_ * N_ * 2); // (B,H,F,N) (reused)
    float*  esrc    = (float*) alloc((size_t)B_ * H_ * N_ * 4);
    float*  edst    = (float*) alloc((size_t)B_ * H_ * N_ * 4);
    bf16_t* attnout = (bf16_t*)alloc((size_t)B_ * N_ * HF_ * 2);     // (B*N, 256) (reused)
    float*  z       = (float*) alloc((size_t)B_ * N_ * D_ * 4);      // (B*N, 128)
    bf16_t* zbf     = (bf16_t*)alloc((size_t)B_ * N_ * D_ * 2);
    bf16_t* zT      = (bf16_t*)alloc((size_t)B_ * D_ * N_ * 2);      // (B, 128, N)
    float*  logits  = (float*) alloc((size_t)B_ * N_ * S_ * 4);
    bf16_t* sT      = (bf16_t*)alloc((size_t)B_ * S_ * N_ * 2);      // (B, 128, N)

    const int TPB = 256;
    // ---- weight / input conversion ----
    {
        int n = B_ * N_ * D_;
        cvt_bf16_kernel<<<(n + TPB - 1) / TPB, TPB, 0, stream>>>(x, xbf, n);
    }
    {
        int n = H_ * D_ * F_;   // W: (H, D, F) -> WT: (H, F, D)
        cvt_transpose_kernel<<<(n + TPB - 1) / TPB, TPB, 0, stream>>>(W_e, WT_e, D_, F_, n);
        cvt_transpose_kernel<<<(n + TPB - 1) / TPB, TPB, 0, stream>>>(W_a, WT_a, D_, F_, n);
    }
    {
        int n = HF_ * D_;       // Wo: (256, 128) -> WoT: (128, 256)
        cvt_transpose_kernel<<<(n + TPB - 1) / TPB, TPB, 0, stream>>>(Wo_e, WoT_e, HF_, D_, n);
        cvt_transpose_kernel<<<(n + TPB - 1) / TPB, TPB, 0, stream>>>(Wo_a, WoT_a, HF_, S_, n);
    }

    const int gridRows = B_ * (N_ / 16);        // 1024
    const int nBHN     = B_ * H_ * N_;

    // ================= layer 1 =================
    proj_kernel<<<gridRows, 128, 0, stream>>>(xbf, WT_e, hT);
    edge_scores_kernel<<<(nBHN + TPB - 1) / TPB, TPB, 0, stream>>>(hT, a_src_e, a_dst_e, esrc, edst);
    attention_kernel<<<gridRows, 128, 0, stream>>>(adj, t, esrc, edst, hT, attnout);
    out_proj_kernel<<<gridRows, 256, 0, stream>>>(attnout, WoT_e, z, zbf, zT, 1);

    // ================= layer 2 =================
    proj_kernel<<<gridRows, 128, 0, stream>>>(zbf, WT_a, hT);
    edge_scores_kernel<<<(nBHN + TPB - 1) / TPB, TPB, 0, stream>>>(hT, a_src_a, a_dst_a, esrc, edst);
    attention_kernel<<<gridRows, 128, 0, stream>>>(adj, t, esrc, edst, hT, attnout);
    out_proj_kernel<<<gridRows, 256, 0, stream>>>(attnout, WoT_a, logits, nullptr, nullptr, 0);

    // ================= heads =================
    softmax_s_kernel<<<B_ * N_, 128, 0, stream>>>(logits, s_out, sT);
    xparent_kernel<<<B_ * (S_ / 16), 256, 0, stream>>>(sT, zT, xp_out);
    tparent_kernel<<<B_, 128, 0, stream>>>(t, s_out, tp_out);
}
